// GradientModel_48198122995732
// MI455X (gfx1250) — matrix operand order repaired and checked by reading, not compiled, not run on hardware
//
#include <hip/hip_runtime.h>
#include <hip/hip_bf16.h>
#include <cstdint>

// ---------------------------------------------------------------------------
// CDNA5 (gfx1250) GNN forward: fused WMMA GEMMs + L2-resident edge scatter.
// All matrix math via v_wmma_f32_16x16x32_bf16 (wave32, 16x16 C tiles, f32 acc).
// ---------------------------------------------------------------------------

typedef __attribute__((ext_vector_type(16))) __bf16 v16bf;
typedef __attribute__((ext_vector_type(8)))  float  v8f;

#define DEV __device__ __forceinline__

DEV unsigned fenc(float f) {
  unsigned u = __float_as_uint(f);
  return (u & 0x80000000u) ? ~u : (u | 0x80000000u);
}
DEV float fdec(unsigned u) {
  return (u & 0x80000000u) ? __uint_as_float(u & 0x7FFFFFFFu) : __uint_as_float(~u);
}
DEV float lrelu(float x) { return x > 0.f ? x : 0.2f * x; }

// ---------------------------------------------------------------------------
// Pack a row-major KxN f32 weight into WMMA B fragments (bf16, contiguous k order, same as the A load):
// per (kt,nt) 32x16 tile, lane = {Khalf(1b), N(4b)}, element e -> K = kh*16+e.
// ---------------------------------------------------------------------------
__global__ void pack_b(const float* __restrict__ W, int K, int Nn,
                       __bf16* __restrict__ pb, int ntTotal, int ntOffset) {
  int lane = threadIdx.x & 31;
  int t = (blockIdx.x * blockDim.x + threadIdx.x) >> 5;  // tile index (wave-uniform)
  int ntL = Nn >> 4, ntK = K >> 5;
  if (t >= ntK * ntL) return;
  int kt = t / ntL, nt = t - kt * ntL;
  int kbase = kt * 32 + (lane >> 4) * 16;
  int n = (nt << 4) + (lane & 15);
  v16bf v;
#pragma unroll
  for (int e = 0; e < 16; ++e) v[e] = (__bf16)W[(size_t)(kbase + e) * Nn + n];
  ((v16bf*)pb)[(size_t)(kt * ntTotal + ntOffset + nt) * 32 + lane] = v;
}

__global__ void build_bias96(const float* __restrict__ b_m1, float* __restrict__ out) {
  int t = threadIdx.x;
  if (t < 96) out[t] = (t < 64) ? b_m1[t] : 0.f;
}

// ---------------------------------------------------------------------------
// Generic WMMA GEMM: C[M x N] = act(A_f32) @ B_packed_bf16 (+bias, +relu-store)
// BN template flag fuses BatchNorm+ReLU into the A-operand load:
//   a = max(0, a*scale[k] + shift[k])
// One wave computes a 16 x (16*CT) tile; K-loop over 32-wide steps.
// All CT B-fragments are loaded before the CT WMMAs so the scheduler can
// clause the loads and overlap them with WMMA issue (partial loadcnt waits).
// ---------------------------------------------------------------------------
template <int CT, bool BN>
__global__ void gemm_wmma(const float* __restrict__ A, int lda, int Mtiles,
                          const __bf16* __restrict__ pb, int ntN, int ntK,
                          float* __restrict__ C, int ldc,
                          const float* __restrict__ bias,
                          const float* __restrict__ aScale,
                          const float* __restrict__ aShift, int reluStore) {
  int lane = threadIdx.x & 31;
  int wave = (blockIdx.x * blockDim.x + threadIdx.x) >> 5;
  int colGroups = ntN / CT;
  int rowTile = wave / colGroups;
  int cg = wave - rowTile * colGroups;
  if (rowTile >= Mtiles) return;  // wave-uniform: EXEC stays all-ones for WMMA

  int mlane = lane & 15, hs = lane >> 4;
  const float* arow = A + (size_t)(rowTile * 16 + mlane) * lda;
  v8f acc[CT] = {};
  const v16bf* pbv = (const v16bf*)pb;

  for (int kt = 0; kt < ntK; ++kt) {
    // ---- B fragments: batch the loads so they land in one clause ----
    const v16bf* bp = pbv + ((size_t)kt * ntN + cg * CT) * 32 + lane;
    v16bf bfrag[CT];
#pragma unroll
    for (int c = 0; c < CT; ++c) bfrag[c] = bp[c * 32];

    // ---- A fragment (f32 -> bf16 in-register, optional fused BN+ReLU) ----
    // lane layout: elems 0..7 -> K = ka.., elems 8..15 -> K = ka+8..
    int ka = kt * 32 + hs * 16;
    v16bf af;
#pragma unroll
    for (int j = 0; j < 8; ++j) {
      int k0 = ka + j, k1 = ka + 8 + j;
      float v0 = arow[k0], v1 = arow[k1];
      if (BN) {
        v0 = fmaxf(v0 * aScale[k0] + aShift[k0], 0.f);
        v1 = fmaxf(v1 * aScale[k1] + aShift[k1], 0.f);
      }
      af[j] = (__bf16)v0;
      af[j + 8] = (__bf16)v1;
    }

#pragma unroll
    for (int c = 0; c < CT; ++c) {
      acc[c] = __builtin_amdgcn_wmma_f32_16x16x32_bf16(
          false, af, false, bfrag[c], (short)0, acc[c], false, false);
    }
  }

  // C/D layout: VGPR v, lanes 0-15 -> M=v, lanes 16-31 -> M=v+8; N = lane&15
  int n0 = cg * CT * 16 + mlane;
#pragma unroll
  for (int c = 0; c < CT; ++c) {
    int n = n0 + c * 16;
    float bv = bias ? bias[n] : 0.f;
#pragma unroll
    for (int v = 0; v < 8; ++v) {
      size_t m = (size_t)rowTile * 16 + hs * 8 + v;
      float val = acc[c][v] + bv;
      if (reluStore) val = fmaxf(val, 0.f);
      C[m * ldc + n] = val;
    }
  }
}

// ---------------------------------------------------------------------------
// BatchNorm training-mode stats (biased var) -> per-channel scale/shift
// ---------------------------------------------------------------------------
__global__ void bn_stats(const float* __restrict__ H, int ld, int M,
                         const float* __restrict__ g, const float* __restrict__ be,
                         float* __restrict__ scale, float* __restrict__ shift) {
  int c = blockIdx.x;
  float s = 0.f, q = 0.f;
  for (int i = threadIdx.x; i < M; i += blockDim.x) {
    float v = H[(size_t)i * ld + c];
    s += v;
    q += v * v;
  }
  __shared__ float ss[256], sq[256];
  ss[threadIdx.x] = s;
  sq[threadIdx.x] = q;
  __syncthreads();
  for (int o = 128; o > 0; o >>= 1) {
    if (threadIdx.x < o) {
      ss[threadIdx.x] += ss[threadIdx.x + o];
      sq[threadIdx.x] += sq[threadIdx.x + o];
    }
    __syncthreads();
  }
  if (threadIdx.x == 0) {
    float m = ss[0] / (float)M;
    float var = sq[0] / (float)M - m * m;
    float sc = g[c] * rsqrtf(var + 1e-5f);
    scale[c] = sc;
    shift[c] = be[c] - m * sc;
  }
}

// ---------------------------------------------------------------------------
// GAT helpers
// ---------------------------------------------------------------------------
__global__ void gat_scores(const float* __restrict__ H, int ldh,
                           const float* __restrict__ a_src, const float* __restrict__ a_dst,
                           float* __restrict__ S, float* __restrict__ D, int N) {
  __shared__ float sa[32], sd[32];
  if (threadIdx.x < 32) {
    sa[threadIdx.x] = a_src[threadIdx.x];
    sd[threadIdx.x] = a_dst[threadIdx.x];
  }
  __syncthreads();
  int i = blockIdx.x * blockDim.x + threadIdx.x;
  if (i >= N) return;
  const float* h = H + (size_t)i * ldh;
  float s = 0.f, d = 0.f;
#pragma unroll
  for (int f = 0; f < 32; ++f) {
    float v = h[f];
    s += v * sa[f];
    d += v * sd[f];
  }
  S[i] = s;
  D[i] = d;
}

__global__ void gat_init(unsigned* __restrict__ mx, float* __restrict__ den,
                         float* __restrict__ num, int N) {
  int t = blockIdx.x * blockDim.x + threadIdx.x;
  if (t < N * 32) num[t] = 0.f;
  if (t < N) {
    mx[t] = fenc(-3.0e38f);
    den[t] = 0.f;
  }
}

// edge pass 1: segment-max of leaky(e) over dst (self-loops in tail range)
__global__ void gat_pass_max(const int* __restrict__ src, const int* __restrict__ dst,
                             int E, int N, const float* __restrict__ S,
                             const float* __restrict__ D, unsigned* __restrict__ mx) {
  int idx = blockIdx.x * blockDim.x + threadIdx.x;
  if (idx >= E + N) return;
  int s_, d_;
  if (idx < E) { s_ = src[idx]; d_ = dst[idx]; }
  else         { s_ = d_ = idx - E; }
  atomicMax(&mx[d_], fenc(lrelu(S[s_] + D[d_])));
}

// edge pass 2: den += e^x ; num += e^x * h[src]   (softmax div deferred to fin)
__global__ void gat_pass_sum(const int* __restrict__ src, const int* __restrict__ dst,
                             int E, int N, const float* __restrict__ S,
                             const float* __restrict__ D, const unsigned* __restrict__ mx,
                             const float* __restrict__ H, int ldh,
                             float* __restrict__ den, float* __restrict__ num) {
  int idx = blockIdx.x * blockDim.x + threadIdx.x;
  if (idx >= E + N) return;
  int s_, d_;
  if (idx < E) { s_ = src[idx]; d_ = dst[idx]; }
  else         { s_ = d_ = idx - E; }
  float e = lrelu(S[s_] + D[d_]);
  float ex = __expf(e - fdec(mx[d_]));
  atomicAdd(&den[d_], ex);
  const float* h = H + (size_t)s_ * ldh;
  float* nm = num + (size_t)d_ * 32;
#pragma unroll
  for (int f = 0; f < 32; ++f) atomicAdd(&nm[f], ex * h[f]);
}

__global__ void gat_fin(const float* __restrict__ num, const float* __restrict__ den,
                        const float* __restrict__ b, float* __restrict__ out, int N) {
  int t = blockIdx.x * blockDim.x + threadIdx.x;
  if (t >= N * 32) return;
  int i = t >> 5, f = t & 31;
  out[t] = fmaxf(num[t] / den[i] + b[f], 0.f);  // relu(_gat(...))
}

// z = 0.5*bn_relu(hm3) + 0.5*hgf
__global__ void blend_k(const float* __restrict__ hm3, const float* __restrict__ sc,
                        const float* __restrict__ sh, const float* __restrict__ hgf,
                        float* __restrict__ z, int N) {
  int t = blockIdx.x * blockDim.x + threadIdx.x;
  if (t >= N * 32) return;
  int f = t & 31;
  float r = fmaxf(hm3[t] * sc[f] + sh[f], 0.f);
  z[t] = 0.5f * r + 0.5f * hgf[t];
}

// ---------------------------------------------------------------------------
extern "C" void kernel_launch(void* const* d_in, const int* in_sizes, int n_in,
                              void* d_out, int out_size, void* d_ws, size_t ws_size,
                              hipStream_t stream) {
  const float* x     = (const float*)d_in[0];
  const int*   ei    = (const int*)d_in[1];
  const float* W_g1  = (const float*)d_in[2];
  const float* as_g1 = (const float*)d_in[3];
  const float* ad_g1 = (const float*)d_in[4];
  const float* b_g1  = (const float*)d_in[5];
  const float* W_g2  = (const float*)d_in[6];
  const float* as_g2 = (const float*)d_in[7];
  const float* ad_g2 = (const float*)d_in[8];
  const float* b_g2  = (const float*)d_in[9];
  const float* W_gf  = (const float*)d_in[10];
  const float* b_gf  = (const float*)d_in[11];
  const float* W_m1  = (const float*)d_in[12];
  const float* b_m1  = (const float*)d_in[13];
  const float* g_m1  = (const float*)d_in[14];
  const float* be_m1 = (const float*)d_in[15];
  const float* W_m2  = (const float*)d_in[16];
  const float* b_m2  = (const float*)d_in[17];
  const float* g_m2  = (const float*)d_in[18];
  const float* be_m2 = (const float*)d_in[19];
  const float* W_m3  = (const float*)d_in[20];
  const float* b_m3  = (const float*)d_in[21];
  const float* g_m3  = (const float*)d_in[22];
  const float* be_m3 = (const float*)d_in[23];
  const float* W_f   = (const float*)d_in[24];
  const float* b_f   = (const float*)d_in[25];
  float* out = (float*)d_out;

  const int N = in_sizes[0] / 256;  // 50000 (divisible by 16)
  const int E = in_sizes[1] / 2;    // 1600000
  const int Mt = N / 16;            // 3125 row tiles
  const int* src = ei;
  const int* dst = ei + E;

  // ---- workspace carve (256B aligned) ----
  char* p = (char*)d_ws;
  auto carve = [&](size_t bytes) -> void* {
    void* r = (void*)p;
    p += (bytes + 255) & ~(size_t)255;
    return r;
  };
  __bf16* pb1   = (__bf16*)carve((size_t)8 * 6 * 512 * 2);   // [W_m1 | W_g1] 256x96
  __bf16* pb_m2 = (__bf16*)carve((size_t)2 * 4 * 512 * 2);   // 64x64
  __bf16* pb_m3 = (__bf16*)carve((size_t)2 * 2 * 512 * 2);   // 64x32
  __bf16* pb_g2 = (__bf16*)carve((size_t)1 * 2 * 512 * 2);   // 32x32
  __bf16* pb_gf = (__bf16*)carve((size_t)1 * 2 * 512 * 2);   // 32x32
  __bf16* pb_f  = (__bf16*)carve((size_t)1 * 16 * 512 * 2);  // 32x256
  float* bias96 = (float*)carve(96 * 4);
  float* hbuf1  = (float*)carve((size_t)N * 96 * 4);  // cols 0..63: m1 raw, 64..95: g1
  float* sc1 = (float*)carve(64 * 4); float* sh1 = (float*)carve(64 * 4);
  float* sc2 = (float*)carve(64 * 4); float* sh2 = (float*)carve(64 * 4);
  float* sc3 = (float*)carve(64 * 4); float* sh3 = (float*)carve(64 * 4);
  float*    sA  = (float*)carve((size_t)N * 4);
  float*    dA  = (float*)carve((size_t)N * 4);
  unsigned* mx  = (unsigned*)carve((size_t)N * 4);
  float*    den = (float*)carve((size_t)N * 4);
  float*    num = (float*)carve((size_t)N * 32 * 4);
  float* hg1   = (float*)carve((size_t)N * 32 * 4);  // GAT1 output
  float* h2raw = (float*)carve((size_t)N * 32 * 4);  // hg1 @ W_g2
  float* hg2   = (float*)carve((size_t)N * 32 * 4);  // GAT2 output
  float* hgf   = (float*)carve((size_t)N * 32 * 4);  // hg2 @ W_gf + b_gf
  float* hm2   = (float*)carve((size_t)N * 64 * 4);  // MLP layer2 raw
  float* hm3   = (float*)carve((size_t)N * 32 * 4);  // MLP layer3 raw
  float* zb    = (float*)carve((size_t)N * 32 * 4);  // blended latent

  auto cdiv = [](long a, long b) { return (int)((a + b - 1) / b); };

  // ---- weight packing (tiny, L2 resident) ----
  pack_b<<<cdiv(8 * 4 * 32, 256), 256, 0, stream>>>(W_m1, 256, 64, pb1, 6, 0);
  pack_b<<<cdiv(8 * 2 * 32, 256), 256, 0, stream>>>(W_g1, 256, 32, pb1, 6, 4);
  pack_b<<<1, 256, 0, stream>>>(W_m2, 64, 64, pb_m2, 4, 0);
  pack_b<<<1, 128, 0, stream>>>(W_m3, 64, 32, pb_m3, 2, 0);
  pack_b<<<1, 64, 0, stream>>>(W_g2, 32, 32, pb_g2, 2, 0);
  pack_b<<<1, 64, 0, stream>>>(W_gf, 32, 32, pb_gf, 2, 0);
  pack_b<<<cdiv(1 * 16 * 32, 256), 256, 0, stream>>>(W_f, 32, 256, pb_f, 16, 0);
  build_bias96<<<1, 96, 0, stream>>>(b_m1, bias96);

  // ---- fused GEMM1: x @ [W_m1|W_g1] -> hbuf1 (one pass over x) ----
  gemm_wmma<6, false><<<cdiv((long)Mt, 4), 128, 0, stream>>>(
      x, 256, Mt, pb1, 6, 8, hbuf1, 96, bias96, nullptr, nullptr, 0);

  // ---- MLP layer1 stats (BN applied lazily at next GEMM's A-load) ----
  bn_stats<<<64, 256, 0, stream>>>(hbuf1, 96, N, g_m1, be_m1, sc1, sh1);

  // ---- GAT layer 1 on hbuf1 cols 64..95 ----
  gat_scores<<<cdiv(N, 256), 256, 0, stream>>>(hbuf1 + 64, 96, as_g1, ad_g1, sA, dA, N);
  gat_init<<<cdiv((long)N * 32, 256), 256, 0, stream>>>(mx, den, num, N);
  gat_pass_max<<<cdiv((long)E + N, 256), 256, 0, stream>>>(src, dst, E, N, sA, dA, mx);
  gat_pass_sum<<<cdiv((long)E + N, 256), 256, 0, stream>>>(src, dst, E, N, sA, dA, mx,
                                                           hbuf1 + 64, 96, den, num);
  gat_fin<<<cdiv((long)N * 32, 256), 256, 0, stream>>>(num, den, b_g1, hg1, N);

  // ---- GAT layer 2 ----
  gemm_wmma<2, false><<<cdiv((long)Mt, 4), 128, 0, stream>>>(
      hg1, 32, Mt, pb_g2, 2, 1, h2raw, 32, nullptr, nullptr, nullptr, 0);
  gat_scores<<<cdiv(N, 256), 256, 0, stream>>>(h2raw, 32, as_g2, ad_g2, sA, dA, N);
  gat_init<<<cdiv((long)N * 32, 256), 256, 0, stream>>>(mx, den, num, N);
  gat_pass_max<<<cdiv((long)E + N, 256), 256, 0, stream>>>(src, dst, E, N, sA, dA, mx);
  gat_pass_sum<<<cdiv((long)E + N, 256), 256, 0, stream>>>(src, dst, E, N, sA, dA, mx,
                                                           h2raw, 32, den, num);
  gat_fin<<<cdiv((long)N * 32, 256), 256, 0, stream>>>(num, den, b_g2, hg2, N);

  // ---- GAT final linear: hgf = hg2 @ W_gf + b_gf ----
  gemm_wmma<2, false><<<cdiv((long)Mt, 4), 128, 0, stream>>>(
      hg2, 32, Mt, pb_gf, 2, 1, hgf, 32, b_gf, nullptr, nullptr, 0);

  // ---- MLP layer2: A = bn_relu(hbuf1[:, :64]) fused into A-load ----
  gemm_wmma<4, true><<<cdiv((long)Mt, 4), 128, 0, stream>>>(
      hbuf1, 96, Mt, pb_m2, 4, 2, hm2, 64, b_m2, sc1, sh1, 0);
  bn_stats<<<64, 256, 0, stream>>>(hm2, 64, N, g_m2, be_m2, sc2, sh2);

  // ---- MLP layer3 ----
  gemm_wmma<2, true><<<cdiv((long)Mt, 4), 128, 0, stream>>>(
      hm2, 64, Mt, pb_m3, 2, 2, hm3, 32, b_m3, sc2, sh2, 0);
  bn_stats<<<32, 256, 0, stream>>>(hm3, 32, N, g_m3, be_m3, sc3, sh3);

  // ---- blend + final GEMM (relu on store) ----
  blend_k<<<cdiv((long)N * 32, 256), 256, 0, stream>>>(hm3, sc3, sh3, hgf, zb, N);
  gemm_wmma<8, false><<<cdiv((long)Mt * 2, 4), 128, 0, stream>>>(
      zb, 32, Mt, pb_f, 16, 1, out, 256, b_f, nullptr, nullptr, 1);
}
